// MultiHeadAttention_81999515616076
// MI455X (gfx1250) — matrix operand order repaired and checked by reading, not compiled, not run on hardware
//
#include <hip/hip_runtime.h>

#define DEV __device__ __forceinline__

typedef __attribute__((ext_vector_type(16))) _Float16 v16h;
typedef __attribute__((ext_vector_type(8)))  float    v8f;

union Frag {
  v16h v;
  unsigned int u[8];
};

DEV unsigned short f2h(float f) {
  _Float16 h = (_Float16)f;
  unsigned short u;
  __builtin_memcpy(&u, &h, 2);
  return u;
}

DEV v8f wmma16(const Frag& a, const Frag& b, v8f c) {
  return __builtin_amdgcn_wmma_f32_16x16x32_f16(false, a.v, false, b.v,
                                                (short)0, c, false, false);
}

// A-fragment 16x32 f16 from row-major src [row][col], ld in elements.
// Lane layout: row = lane%16; lanes 0-15 hold K=0..15, lanes 16-31 K=16..31,
// consecutive K-pairs per VGPR.
DEV Frag load_a(const unsigned short* src, int row0, int col0, int ld, int lane) {
  Frag f;
  int l16 = lane & 15, half = lane >> 4;
  const unsigned short* p = src + (size_t)(row0 + l16) * ld + col0 + 16 * half;
#pragma unroll
  for (int j = 0; j < 8; ++j) {
    f.u[j] = *(const unsigned int*)(p + 2 * j);
  }
  return f;
}

// B-fragment 32x16 f16 where element (k, n) = src[(n0+n)*ld + k0 + k].
// Lane layout: col n = lane%16; lanes 0-15 hold K=0..15, lanes 16-31 K=16..31,
// consecutive K-pairs per VGPR.
DEV Frag load_bt(const unsigned short* src, int n0, int k0, int ld, int lane) {
  Frag f;
  int l16 = lane & 15, half = lane >> 4;
  const unsigned short* p = src + (size_t)(n0 + l16) * ld + k0 + 16 * half;
#pragma unroll
  for (int j = 0; j < 8; ++j)
    f.u[j] = *(const unsigned int*)(p + 2 * j);
  return f;
}

// ---------------------------------------------------------------------------
// Weight transpose + f32->f16 convert: wqkv_t[c][k] (c<1024: W_q, else W_kv),
// wout_t[c][k].
__global__ void convert_w(const float* __restrict__ Wq, const float* __restrict__ Wkv,
                          const float* __restrict__ Wo,
                          unsigned short* wqkv_t, unsigned short* wout_t) {
  size_t i = (size_t)blockIdx.x * 256 + threadIdx.x;
  const size_t NQKV = (size_t)3072 * 1024;
  if (i < NQKV) {
    int c = (int)(i >> 10), k = (int)(i & 1023);
    float v = (c < 1024) ? Wq[(size_t)k * 1024 + c]
                         : Wkv[(size_t)k * 2048 + (c - 1024)];
    wqkv_t[i] = f2h(v);
  } else {
    size_t j = i - NQKV;
    int c = (int)(j >> 10), k = (int)(j & 1023);
    wout_t[j] = f2h(Wo[(size_t)k * 1024 + c]);
  }
}

// ---------------------------------------------------------------------------
// Row LayerNorm (dim elements), output f16 (H16OUT) or f32.
template <bool H16OUT>
__global__ void ln_kernel(const float* __restrict__ x, const float* __restrict__ g,
                          const float* __restrict__ bb, void* out, int dim) {
  int row = blockIdx.x;
  const float* xr = x + (size_t)row * dim;
  __shared__ float redS[8], redS2[8], sh[2];
  float s = 0.f, s2 = 0.f;
  for (int i = threadIdx.x; i < dim; i += blockDim.x) {
    float v = xr[i];
    s += v;
    s2 += v * v;
  }
  for (int m = 1; m < 32; m <<= 1) {
    s += __shfl_xor(s, m, 32);
    s2 += __shfl_xor(s2, m, 32);
  }
  if ((threadIdx.x & 31) == 0) {
    redS[threadIdx.x >> 5] = s;
    redS2[threadIdx.x >> 5] = s2;
  }
  __syncthreads();
  if (threadIdx.x == 0) {
    float a = 0.f, a2 = 0.f;
    int nw = blockDim.x >> 5;
    for (int i = 0; i < nw; ++i) { a += redS[i]; a2 += redS2[i]; }
    float mu = a / dim;
    sh[0] = mu;
    sh[1] = rsqrtf(a2 / dim - mu * mu + 1e-5f);
  }
  __syncthreads();
  float mu = sh[0], rs = sh[1];
  for (int i = threadIdx.x; i < dim; i += blockDim.x) {
    float v = (xr[i] - mu) * rs * g[i] + bb[i];
    if (H16OUT)
      ((unsigned short*)out)[(size_t)row * dim + i] = f2h(v);
    else
      ((float*)out)[(size_t)row * dim + i] = v;
  }
}

// ---------------------------------------------------------------------------
// Context path: LN(c_emb) @ W_ctx + b_ctx -> k_ext rows 1..128 (row-major d),
// v_extT cols 1..128 ([d][key] layout). key 0 = null_kv, keys 129..159 pad.
__global__ void __launch_bounds__(128)
ctx_kernel(const float* __restrict__ c_emb, const float* __restrict__ g,
           const float* __restrict__ bb, const float* __restrict__ Wc,
           const float* __restrict__ bc, unsigned short* kext,
           unsigned short* vextT) {
  int row = blockIdx.x;  // b*128 + m
  int b = row >> 7, m = row & 127;
  const float* xr = c_emb + (size_t)row * 768;
  __shared__ float xn[768];
  __shared__ float redS[4], redS2[4], sh2[2];
  float s = 0.f, s2 = 0.f;
  for (int i = threadIdx.x; i < 768; i += 128) {
    float v = xr[i];
    s += v;
    s2 += v * v;
  }
  for (int msk = 1; msk < 32; msk <<= 1) {
    s += __shfl_xor(s, msk, 32);
    s2 += __shfl_xor(s2, msk, 32);
  }
  if ((threadIdx.x & 31) == 0) {
    redS[threadIdx.x >> 5] = s;
    redS2[threadIdx.x >> 5] = s2;
  }
  __syncthreads();
  if (threadIdx.x == 0) {
    float a = redS[0] + redS[1] + redS[2] + redS[3];
    float a2 = redS2[0] + redS2[1] + redS2[2] + redS2[3];
    float mu = a / 768.f;
    sh2[0] = mu;
    sh2[1] = rsqrtf(a2 / 768.f - mu * mu + 1e-5f);
  }
  __syncthreads();
  float mu = sh2[0], rs = sh2[1];
  for (int i = threadIdx.x; i < 768; i += 128)
    xn[i] = (xr[i] - mu) * rs * g[i] + bb[i];
  __syncthreads();
  int col = threadIdx.x;
  float acc = bc[col];
  for (int t = 0; t < 768; ++t) acc = fmaf(xn[t], Wc[t * 128 + col], acc);
  if (col < 64)
    kext[(size_t)(b * 160 + 1 + m) * 64 + col] = f2h(acc);
  else
    vextT[(size_t)(b * 64 + (col - 64)) * 160 + 1 + m] = f2h(acc);
}

__global__ void ctx_pad(const float* __restrict__ null_kv, unsigned short* kext,
                        unsigned short* vextT) {
  int b = blockIdx.x, d = threadIdx.x;  // d in 0..63
  kext[(size_t)(b * 160) * 64 + d] = f2h(null_kv[d]);
  vextT[(size_t)(b * 64 + d) * 160] = f2h(null_kv[64 + d]);
  for (int r = 129; r < 160; ++r) {
    kext[(size_t)(b * 160 + r) * 64 + d] = 0;
    vextT[(size_t)(b * 64 + d) * 160 + r] = 0;
  }
}

// ---------------------------------------------------------------------------
// WMMA GEMM: one wave computes a 64x64 tile of A[M][K] @ Bt[N][K]^T
// (square tile minimizes L2 traffic: bytes ~ M*N*K*2*(1/Tm+1/Tn)).
// MODE 0: QKV projection epilogue (scatter to q/k row-major, v transposed).
// MODE 1: plain f32 store to outf (ld = Ncols).
template <int MODE>
__global__ void __launch_bounds__(32)
gemm_w64(const unsigned short* __restrict__ A, const unsigned short* __restrict__ Bt,
         int K, unsigned short* qb, unsigned short* kb_, unsigned short* vtb,
         float* outf, int Ncols) {
  int lane = threadIdx.x;
  int l16 = lane & 15, half = lane >> 4;
  int N0 = blockIdx.x * 64;
  int M0 = blockIdx.y * 64;
  v8f acc[4][4];
#pragma unroll
  for (int i = 0; i < 4; ++i)
#pragma unroll
    for (int g = 0; g < 4; ++g) acc[i][g] = (v8f){0, 0, 0, 0, 0, 0, 0, 0};

  for (int kk = 0; kk < K; kk += 32) {
    Frag a[4];
#pragma unroll
    for (int rt = 0; rt < 4; ++rt) a[rt] = load_a(A, M0 + 16 * rt, kk, K, lane);
#pragma unroll
    for (int g = 0; g < 4; ++g) {
      Frag b = load_bt(Bt, N0 + 16 * g, kk, K, lane);
#pragma unroll
      for (int rt = 0; rt < 4; ++rt) acc[rt][g] = wmma16(a[rt], b, acc[rt][g]);
    }
  }
#pragma unroll
  for (int rt = 0; rt < 4; ++rt)
#pragma unroll
    for (int g = 0; g < 4; ++g)
#pragma unroll
      for (int r = 0; r < 8; ++r) {
        int row = M0 + 16 * rt + r + 8 * half;
        int col = N0 + 16 * g + l16;
        float val = acc[rt][g][r];
        if (MODE == 0) {
          int b = row >> 11, n = row & 2047;
          if (col < 1024) {
            int h = col >> 6, d = col & 63;
            qb[(size_t)((b * 16 + h) * 2048 + n) * 64 + d] = f2h(val);
          } else if (col < 2048) {
            int c = col - 1024;
            int h = c >> 6, d = c & 63;
            kb_[(size_t)((b * 16 + h) * 2048 + n) * 64 + d] = f2h(val);
          } else {
            int c = col - 2048;
            int h = c >> 6, d = c & 63;
            vtb[(size_t)((b * 16 + h) * 64 + d) * 2048 + n] = f2h(val);
          }
        } else {
          outf[(size_t)row * Ncols + col] = val;
        }
      }
}

// ---------------------------------------------------------------------------
// Flash attention: one wave = (b, h, 32 query rows); streams 2048 self keys +
// 129 ctx keys (padded to 160, masked) in 32-key steps with online softmax.
// 32-query tile: K/V fragments are loaded once and feed 2 row-tiles -> 2x
// arithmetic intensity vs a 16-query tile (K/V stream is the L2-bound term).
struct AttnState {
  float m[2][8], l[2][8];
  v8f acc[2][4];
};

template <bool MASK>
DEV void attn_step(const unsigned short* Kp, const unsigned short* Vt, int ldv,
                   int kb, int nvalid, const Frag aq[2][2], AttnState& st,
                   unsigned short* pshm, int lane) {
  int l16 = lane & 15, half = lane >> 4;
  Frag bk[2][2];
#pragma unroll
  for (int t = 0; t < 2; ++t) {
    bk[t][0] = load_bt(Kp, kb + 16 * t, 0, 64, lane);
    bk[t][1] = load_bt(Kp, kb + 16 * t, 32, 64, lane);
  }
#pragma unroll
  for (int rt = 0; rt < 2; ++rt) {
    float sf[2][8];
#pragma unroll
    for (int t = 0; t < 2; ++t) {
      v8f s = (v8f){0, 0, 0, 0, 0, 0, 0, 0};
      s = wmma16(aq[rt][0], bk[t][0], s);
      s = wmma16(aq[rt][1], bk[t][1], s);
      bool ok = true;
      if (MASK) ok = (kb + 16 * t + l16) < nvalid;
#pragma unroll
      for (int r = 0; r < 8; ++r) sf[t][r] = ok ? s[r] * 0.125f : -1e30f;
    }
#pragma unroll
    for (int r = 0; r < 8; ++r) {
      float mx = fmaxf(sf[0][r], sf[1][r]);
#pragma unroll
      for (int msk = 1; msk < 16; msk <<= 1)
        mx = fmaxf(mx, __shfl_xor(mx, msk, 32));
      float mn = fmaxf(st.m[rt][r], mx);
      float corr = __expf(st.m[rt][r] - mn);
      st.m[rt][r] = mn;
      float p0 = __expf(sf[0][r] - mn);
      float p1 = __expf(sf[1][r] - mn);
      float rs = p0 + p1;
#pragma unroll
      for (int msk = 1; msk < 16; msk <<= 1) rs += __shfl_xor(rs, msk, 32);
      st.l[rt][r] = st.l[rt][r] * corr + rs;
#pragma unroll
      for (int g = 0; g < 4; ++g) st.acc[rt][g][r] *= corr;
      // C-layout (row 16*rt + r + 8*half, col l16) -> LDS row-major 32x32 f16
      pshm[(16 * rt + 8 * half + r) * 32 + l16] = f2h(p0);
      pshm[(16 * rt + 8 * half + r) * 32 + 16 + l16] = f2h(p1);
    }
  }
  __syncthreads();
  Frag pa0 = load_a(pshm, 0, 0, 32, lane);
  Frag pa1 = load_a(pshm, 16, 0, 32, lane);
#pragma unroll
  for (int g = 0; g < 4; ++g) {
    Frag bv = load_bt(Vt, 16 * g, kb, ldv, lane);
    st.acc[0][g] = wmma16(pa0, bv, st.acc[0][g]);
    st.acc[1][g] = wmma16(pa1, bv, st.acc[1][g]);
  }
  __syncthreads();
}

__global__ void __launch_bounds__(32)
attn_kernel(const unsigned short* __restrict__ qbuf,
            const unsigned short* __restrict__ kbuf,
            const unsigned short* __restrict__ vtbuf,
            const unsigned short* __restrict__ kext,
            const unsigned short* __restrict__ vextT, unsigned short* att) {
  __shared__ unsigned short pshm[32 * 32];
  int lane = threadIdx.x;
  int l16 = lane & 15, half = lane >> 4;
  int n0 = blockIdx.x * 32;
  int bh = blockIdx.y;
  int b = bh >> 4, h = bh & 15;
  const unsigned short* qp = qbuf + (size_t)bh * 2048 * 64;
  const unsigned short* kp = kbuf + (size_t)bh * 2048 * 64;
  const unsigned short* vp = vtbuf + (size_t)bh * 64 * 2048;
  const unsigned short* kx = kext + (size_t)b * 160 * 64;
  const unsigned short* vx = vextT + (size_t)b * 64 * 160;

  Frag aq[2][2];
#pragma unroll
  for (int rt = 0; rt < 2; ++rt) {
    aq[rt][0] = load_a(qp, n0 + 16 * rt, 0, 64, lane);
    aq[rt][1] = load_a(qp, n0 + 16 * rt, 32, 64, lane);
  }

  AttnState st;
#pragma unroll
  for (int rt = 0; rt < 2; ++rt)
#pragma unroll
    for (int r = 0; r < 8; ++r) {
      st.m[rt][r] = -1e30f;
      st.l[rt][r] = 0.f;
    }
#pragma unroll
  for (int rt = 0; rt < 2; ++rt)
#pragma unroll
    for (int g = 0; g < 4; ++g) st.acc[rt][g] = (v8f){0, 0, 0, 0, 0, 0, 0, 0};

  for (int kt = 0; kt < 64; ++kt)
    attn_step<false>(kp, vp, 2048, kt * 32, 0, aq, st, pshm, lane);
  for (int kt = 0; kt < 5; ++kt)
    attn_step<true>(kx, vx, 160, kt * 32, 129, aq, st, pshm, lane);

#pragma unroll
  for (int rt = 0; rt < 2; ++rt)
#pragma unroll
    for (int g = 0; g < 4; ++g)
#pragma unroll
      for (int r = 0; r < 8; ++r) {
        int n = n0 + 16 * rt + r + 8 * half;
        float o = st.acc[rt][g][r] / st.l[rt][r];
        att[(size_t)(b * 2048 + n) * 1024 + h * 64 + 16 * g + l16] = f2h(o);
      }
}

// ---------------------------------------------------------------------------
extern "C" void kernel_launch(void* const* d_in, const int* /*in_sizes*/,
                              int /*n_in*/, void* d_out, int /*out_size*/,
                              void* d_ws, size_t /*ws_size*/, hipStream_t stream) {
  const float* x        = (const float*)d_in[0];
  const float* c_emb    = (const float*)d_in[1];
  const float* ln_g     = (const float*)d_in[2];
  const float* ln_b     = (const float*)d_in[3];
  const float* ctx_ln_g = (const float*)d_in[4];
  const float* ctx_ln_b = (const float*)d_in[5];
  const float* W_ctx    = (const float*)d_in[6];
  const float* b_ctx    = (const float*)d_in[7];
  const float* W_q      = (const float*)d_in[8];
  const float* W_kv     = (const float*)d_in[9];
  const float* null_kv  = (const float*)d_in[10];
  const float* W_out    = (const float*)d_in[11];
  const float* out_ln_g = (const float*)d_in[12];
  const float* out_ln_b = (const float*)d_in[13];
  float* out = (float*)d_out;

  char* ws = (char*)d_ws;
  size_t off = 0;
  auto take = [&](size_t bytes) {
    char* p = ws + off;
    off = (off + bytes + 255) & ~(size_t)255;
    return p;
  };
  unsigned short* xn     = (unsigned short*)take((size_t)4096 * 1024 * 2);
  unsigned short* wqkv_t = (unsigned short*)take((size_t)3072 * 1024 * 2);
  unsigned short* wout_t = (unsigned short*)take((size_t)1024 * 1024 * 2);
  unsigned short* qbuf   = (unsigned short*)take((size_t)2 * 16 * 2048 * 64 * 2);
  unsigned short* kbuf   = (unsigned short*)take((size_t)2 * 16 * 2048 * 64 * 2);
  unsigned short* vtbuf  = (unsigned short*)take((size_t)2 * 16 * 2048 * 64 * 2);
  unsigned short* kext   = (unsigned short*)take((size_t)2 * 160 * 64 * 2);
  unsigned short* vextT  = (unsigned short*)take((size_t)2 * 64 * 160 * 2);
  unsigned short* att    = (unsigned short*)take((size_t)4096 * 1024 * 2);
  float* out_pre         = (float*)take((size_t)4096 * 1024 * 4);

  convert_w<<<16384, 256, 0, stream>>>(W_q, W_kv, W_out, wqkv_t, wout_t);
  ln_kernel<true><<<4096, 256, 0, stream>>>(x, ln_g, ln_b, xn, 1024);
  ctx_pad<<<2, 64, 0, stream>>>(null_kv, kext, vextT);
  ctx_kernel<<<256, 128, 0, stream>>>(c_emb, ctx_ln_g, ctx_ln_b, W_ctx, b_ctx,
                                      kext, vextT);
  gemm_w64<0><<<dim3(48, 64), 32, 0, stream>>>(xn, wqkv_t, 1024, qbuf, kbuf,
                                               vtbuf, nullptr, 3072);
  attn_kernel<<<dim3(64, 32), 32, 0, stream>>>(qbuf, kbuf, vtbuf, kext, vextT,
                                               att);
  gemm_w64<1><<<dim3(16, 64), 32, 0, stream>>>(att, wout_t, 1024, nullptr,
                                               nullptr, nullptr, out_pre, 1024);
  ln_kernel<false><<<4096, 256, 0, stream>>>(out_pre, out_ln_g, out_ln_b, out,
                                             1024);
}
